// RNN_22780506538196
// MI455X (gfx1250) — matrix operand order repaired and checked by reading, not compiled, not run on hardware
//
#include <hip/hip_runtime.h>
#include <hip/hip_bf16.h>

// Problem constants (match reference)
#define T_STEPS 100
#define B_SZ    256
#define Q_SZ    2048
#define H_SZ    1024
#define NIN_SZ  (2 * Q_SZ)

typedef __bf16 bf16x16 __attribute__((ext_vector_type(16)));
typedef float  f32x8   __attribute__((ext_vector_type(8)));

union BfVec { bf16x16 v; unsigned int u[8]; };

// round-to-nearest-even f32 -> bf16 (bit pattern)
__device__ __forceinline__ unsigned short f2bf(float f) {
    unsigned int u = __float_as_uint(f);
    u = (u + 0x7FFFu + ((u >> 16) & 1u)) >> 16;
    return (unsigned short)u;
}

// ---------------------------------------------------------------------------
// One-time: convert Wm (f32, HxH row-major) to bf16
// ---------------------------------------------------------------------------
__global__ __launch_bounds__(256) void dkt_cvt_wm(const float* __restrict__ Wm,
                                                  unsigned short* __restrict__ Wmb) {
    int i = blockIdx.x * blockDim.x + threadIdx.x;           // grid covers H*H
    Wmb[i] = f2bf(Wm[i]);
}

// One-time: h0[b][j] = b_start[j]  (zeros @ W_start^T == 0), stored as bf16
__global__ __launch_bounds__(256) void dkt_init_h(const float* __restrict__ b_start,
                                                  unsigned short* __restrict__ hb) {
    int i = blockIdx.x * blockDim.x + threadIdx.x;           // grid covers B*H
    hb[i] = f2bf(b_start[i & (H_SZ - 1)]);
}

// ---------------------------------------------------------------------------
// Recurrent step: h_new = tanh(h_old @ Wm^T + bm + Wx[:, x_idx] + bx)
// Wave32 WMMA bf16: each wave computes a 16(b) x 32(j) strip (2 accumulators),
// looping K over H in chunks of 32 -> v_wmma_f32_16x16x32_bf16.
// Grid: 512 waves = 64 blocks x 8 waves.
// ---------------------------------------------------------------------------
__global__ __launch_bounds__(256) void dkt_step(
    const unsigned short* __restrict__ hin,   // [B,H] bf16
    unsigned short*       __restrict__ hout,  // [B,H] bf16
    float*                __restrict__ hf,    // [B,H] f32 master
    const unsigned short* __restrict__ Wmb,   // [H,H] bf16 (row j, col k)
    const float*          __restrict__ bm,    // [H]
    const float*          __restrict__ Wx,    // [H,NIN] row-major
    const float*          __restrict__ bx,    // [H]
    const int*            __restrict__ xid)   // x_idx + t*B
{
    const int lane = threadIdx.x & 31;
    const int wid  = (blockIdx.x * blockDim.x + threadIdx.x) >> 5;  // 0..511
    const int b0   = (wid & 15) << 4;          // 16 b-tiles
    const int j0   = (wid >> 4) << 5;          // 32 j-strips of width 32

    const int half = lane >> 4;                // lane group
    const int lq   = lane & 15;

    f32x8 acc0 = {};
    f32x8 acc1 = {};

    const unsigned short* arow  = hin + (size_t)(b0 + lq) * H_SZ;          // A row (M = lq)
    const unsigned short* brow0 = Wmb + (size_t)(j0 + lq) * H_SZ;          // B col N=lq, tile 0
    const unsigned short* brow1 = Wmb + (size_t)(j0 + 16 + lq) * H_SZ;     // B col N=lq, tile 1

    for (int k0 = 0; k0 < H_SZ; k0 += 32) {
        BfVec a, bA, bB;
        // A: lanes 0-15 hold K=0..15 in V0..7 (2 per VGPR),
        //    lanes 16-31 hold K=16..31 (same k order as B).
        #pragma unroll
        for (int i = 0; i < 8; ++i)
            a.u[i] = *(const unsigned int*)(arow + k0 + 16 * half + 2 * i);

        // B: 32x16 layout — lanes 0-15 hold K=0..15 in V0..7 (2 per VGPR),
        //    lanes 16-31 hold K=16..31.  B[k][n] = Wm[j][k] -> contiguous 32b loads.
        const int kb = k0 + 16 * half;
        #pragma unroll
        for (int i = 0; i < 8; ++i) {
            bA.u[i] = *(const unsigned int*)(brow0 + kb + 2 * i);
            bB.u[i] = *(const unsigned int*)(brow1 + kb + 2 * i);
        }

        acc0 = __builtin_amdgcn_wmma_f32_16x16x32_bf16(false, a.v, false, bA.v,
                                                       (short)0, acc0, false, false);
        acc1 = __builtin_amdgcn_wmma_f32_16x16x32_bf16(false, a.v, false, bB.v,
                                                       (short)0, acc1, false, false);
    }

    // Epilogue: + bm[j] + Wx[j, x_idx[b]] + bx[j], tanh, store f32 + bf16.
    // C/D layout: element r of v8f is row M = r + 8*half, column N = lq.
    int xs[8];
    #pragma unroll
    for (int r = 0; r < 8; ++r)
        xs[r] = xid[b0 + r + 8 * half];

    #pragma unroll
    for (int jt = 0; jt < 2; ++jt) {
        const int j = j0 + jt * 16 + lq;
        const float bias  = bm[j] + bx[j];
        const float* wxr  = Wx + (size_t)j * NIN_SZ;
        f32x8 acc = jt ? acc1 : acc0;
        #pragma unroll
        for (int r = 0; r < 8; ++r) {
            const int b = b0 + r + 8 * half;
            float v = acc[r] + bias + wxr[xs[r]];
            float h = tanhf(v);
            hf[(size_t)b * H_SZ + j]   = h;
            hout[(size_t)b * H_SZ + j] = f2bf(h);
        }
    }
}

// ---------------------------------------------------------------------------
// Per-step output: logits[b] = <h[b,:], Wy[y_idx[b],:]> + by[y_idx]
// pred = sigmoid(logit); loss = stable BCE via log_sigmoid.
// One wave per b (wave32), deterministic shuffle reduction.
// ---------------------------------------------------------------------------
__global__ __launch_bounds__(256) void dkt_out(
    const float* __restrict__ hf,    // [B,H] f32
    const int*   __restrict__ yid,   // y_idx + t*B
    const float* __restrict__ tr,    // truth + t*B
    const float* __restrict__ Wy,    // [Q,H]
    const float* __restrict__ by,    // [Q]
    float*       __restrict__ pred,  // d_out + t*B
    float*       __restrict__ loss)  // d_out + T*B + t*B
{
    const int lane = threadIdx.x & 31;
    const int b    = (blockIdx.x * blockDim.x + threadIdx.x) >> 5;  // 0..255
    const int yi   = yid[b];
    const float* hrow = hf + (size_t)b * H_SZ;
    const float* wrow = Wy + (size_t)yi * H_SZ;

    float s = 0.f;
    #pragma unroll 8
    for (int j = lane; j < H_SZ; j += 32)
        s = fmaf(hrow[j], wrow[j], s);

    #pragma unroll
    for (int off = 16; off >= 1; off >>= 1)
        s += __shfl_xor(s, off, 32);

    if (lane == 0) {
        float x = s + by[yi];
        float p = 1.f / (1.f + __expf(-x));
        // log_sigmoid(x) stable both signs
        float lsp = (x >= 0.f) ? -log1pf(__expf(-x)) : (x - log1pf(__expf(x)));
        float lsn = (x >= 0.f) ? (-x - log1pf(__expf(-x))) : -log1pf(__expf(x));
        float t = tr[b];
        pred[b] = p;
        loss[b] = -(t * lsp + (1.f - t) * lsn);
    }
}

// ---------------------------------------------------------------------------
extern "C" void kernel_launch(void* const* d_in, const int* in_sizes, int n_in,
                              void* d_out, int out_size, void* d_ws, size_t ws_size,
                              hipStream_t stream) {
    (void)in_sizes; (void)n_in; (void)out_size; (void)ws_size;

    const int*   x_idx   = (const int*)  d_in[0];   // [T,B]
    const int*   y_idx   = (const int*)  d_in[1];   // [T,B]
    const float* truth   = (const float*)d_in[2];   // [T,B]
    const float* Wm      = (const float*)d_in[3];   // [H,H]
    const float* bm      = (const float*)d_in[4];   // [H]
    const float* Wx      = (const float*)d_in[5];   // [H,NIN]
    const float* bx      = (const float*)d_in[6];   // [H]
    const float* Wy      = (const float*)d_in[7];   // [Q,H]
    const float* by      = (const float*)d_in[8];   // [Q]
    /* d_in[9] = W_start: multiplied by zeros -> unused */
    const float* b_start = (const float*)d_in[10];  // [H]

    // Workspace layout (all we need: ~4 MB)
    unsigned short* Wmb = (unsigned short*)d_ws;                     // H*H bf16
    unsigned short* hb0 = Wmb + (size_t)H_SZ * H_SZ;                 // B*H bf16
    unsigned short* hb1 = hb0 + (size_t)B_SZ * H_SZ;                 // B*H bf16
    float*          hf  = (float*)(hb1 + (size_t)B_SZ * H_SZ);       // B*H f32

    float* pred = (float*)d_out;                 // [T,B]
    float* loss = pred + (size_t)T_STEPS * B_SZ; // [T,B]

    dkt_cvt_wm<<<(H_SZ * H_SZ) / 256, 256, 0, stream>>>(Wm, Wmb);
    dkt_init_h<<<(B_SZ * H_SZ) / 256, 256, 0, stream>>>(b_start, hb0);

    for (int t = 0; t < T_STEPS; ++t) {
        const unsigned short* hin  = (t & 1) ? hb1 : hb0;
        unsigned short*       hout = (t & 1) ? hb0 : hb1;
        dkt_step<<<64, 256, 0, stream>>>(hin, hout, hf, Wmb, bm, Wx, bx,
                                         x_idx + (size_t)t * B_SZ);
        dkt_out<<<32, 256, 0, stream>>>(hf, y_idx + (size_t)t * B_SZ,
                                        truth + (size_t)t * B_SZ, Wy, by,
                                        pred + (size_t)t * B_SZ,
                                        loss + (size_t)t * B_SZ);
    }
}
